// TemporalAttention_61873298866242
// MI455X (gfx1250) — matrix operand order repaired and checked by reading, not compiled, not run on hardware
//
#include <hip/hip_runtime.h>
#include <hip/hip_bf16.h>

typedef unsigned int  u32;
typedef unsigned short u16;
typedef unsigned long long u64;
typedef __attribute__((ext_vector_type(4)))  u32    v4u;
typedef __attribute__((ext_vector_type(8)))  u32    v8u;
typedef __attribute__((ext_vector_type(16))) __bf16 v16bf;
typedef __attribute__((ext_vector_type(8)))  float  v8f;

#define NHEADS 16
#define BATCH  64
#define TLEN   256
#define DMODEL 1024
#define DHEAD  64
#define HB     (NHEADS * BATCH)          // 1024
#define NEG_BIG (-4294967295.0f)         // -(2^32 - 1)

__device__ __forceinline__ u16 f2bf(float f) {
    u32 u = __builtin_bit_cast(u32, f);
    u += 0x7fffu + ((u >> 16) & 1u);     // round-to-nearest-even
    return (u16)(u >> 16);
}

__device__ __forceinline__ v16bf make_frag(v4u a, v4u b) {
    v8u c;
    c[0] = a[0]; c[1] = a[1]; c[2] = a[2]; c[3] = a[3];
    c[4] = b[0]; c[5] = b[1]; c[6] = b[2]; c[7] = b[3];
    return __builtin_bit_cast(v16bf, c);
}

// Low 32 bits of a generic pointer to __shared__ = LDS byte offset
// (ISA 10.2: LDS aperture addresses truncate to addr[31:0]).
__device__ __forceinline__ u32 lds_off(const void* p) {
    return (u32)(uintptr_t)p;
}

// CDNA5 async global->LDS copy, 16B per lane. INST_OFFSET applies to both the
// LDS and global addresses, so one base pair serves several chunks.
#define ASYNC_LDS_B128(lds, gptr, off_)                                        \
    asm volatile("global_load_async_to_lds_b128 %0, %1, off offset:" #off_     \
                 :: "v"(lds), "v"(gptr) : "memory")

#define WAIT_ASYNCCNT0() asm volatile("s_wait_asynccnt 0" ::: "memory")

#define WMMA_BF16(A, B, C) \
    __builtin_amdgcn_wmma_f32_16x16x32_bf16(false, (A), false, (B), (short)0, (C), false, false)

// ---------------------------------------------------------------------------
// Kernel 1: Wt[w][n][k] (bf16) = W[w][k][n] (f32)   -- transpose + convert
// ---------------------------------------------------------------------------
__global__ __launch_bounds__(256)
void prep_weights(const float* __restrict__ Wq, const float* __restrict__ Wk,
                  const float* __restrict__ Wv, u16* __restrict__ Wt) {
    int idx = blockIdx.x * 256 + threadIdx.x;         // < 3 * 1M
    int w   = idx >> 20;
    int rem = idx & ((1 << 20) - 1);
    int k   = rem >> 10;
    int n   = rem & 1023;
    const float* W = (w == 0) ? Wq : (w == 1) ? Wk : Wv;
    Wt[(size_t)w * (DMODEL * DMODEL) + (size_t)n * DMODEL + k] =
        f2bf(W[(size_t)k * DMODEL + n]);
}

// ---------------------------------------------------------------------------
// Kernel 2: temporal bf16 = x + pos_emb  ([B*T, D])
// ---------------------------------------------------------------------------
__global__ __launch_bounds__(256)
void prep_temporal(const float* __restrict__ x, const float* __restrict__ pos,
                   u16* __restrict__ Tmp) {
    size_t i4  = ((size_t)blockIdx.x * 256 + threadIdx.x) * 4;  // 4 elems/thread
    int row    = (int)(i4 >> 10);
    int d      = (int)(i4 & 1023);
    const float4 xv = *(const float4*)&x[i4];
    const float4 pv = *(const float4*)&pos[(size_t)(row & (TLEN - 1)) * DMODEL + d];
    u32 lo = (u32)f2bf(xv.x + pv.x) | ((u32)f2bf(xv.y + pv.y) << 16);
    u32 hi = (u32)f2bf(xv.z + pv.z) | ((u32)f2bf(xv.w + pv.w) << 16);
    u32* o = (u32*)&Tmp[i4];
    o[0] = lo; o[1] = hi;
}

// ---------------------------------------------------------------------------
// Kernel 3: QKV GEMM   C[16384 x 1024] = Tmp @ W(z)
// WG tile 64(M) x 128(N), 8 waves (2x4), each wave 32x32 (2x2 WMMA accums).
// K-step 64 (8 WMMAs per barrier pair). Tiles staged with
// GLOBAL_LOAD_ASYNC_TO_LDS_B128 (ASYNCcnt) instead of VGPR round-trips.
// Epilogue scatters into head-split layouts:
//   z=0: Q  [hb][t][64]   z=1: K [hb][s][64]   z=2: Vt [hb][64][t]
// ---------------------------------------------------------------------------
__global__ __launch_bounds__(256)
void qkv_gemm(const u16* __restrict__ Tmp, const u16* __restrict__ Wt,
              u16* __restrict__ Q, u16* __restrict__ Kh, u16* __restrict__ Vt) {
    __shared__ __align__(16) u16 As[64][64];    //  8 KB
    __shared__ __align__(16) u16 Bs[128][64];   // 16 KB

    const int tid   = threadIdx.x;
    const int lane  = tid & 31;
    const int wave  = tid >> 5;
    const int hs    = lane >> 4;
    const int ln    = lane & 15;
    const int mBase = blockIdx.y * 64;
    const int nBase = blockIdx.x * 128;
    const int w     = blockIdx.z;
    const u16* Wsel = Wt + (size_t)w * (DMODEL * DMODEL);
    const int waveM = (wave >> 2) * 32;   // 0 / 32
    const int waveN = (wave & 3) * 32;    // 0..96

    // per-thread staging assignments (constant across K loop)
    const int ar = tid >> 2, ac = (tid & 3) * 16;   // A: 32B/thread
    const int bn = tid >> 1, bc = (tid & 1) * 32;   // B: 64B/thread
    const u32 aL = lds_off(&As[ar][ac]);
    const u32 bL = lds_off(&Bs[bn][bc]);

    v8f acc[2][2] = {};

    for (int k0 = 0; k0 < DMODEL; k0 += 64) {
        {   // stage A tile: 64 rows x 64 halfs
            const u16* g = &Tmp[(size_t)(mBase + ar) * DMODEL + k0 + ac];
            ASYNC_LDS_B128(aL, g, 0);
            ASYNC_LDS_B128(aL, g, 16);
        }
        {   // stage Bt tile: 128 rows x 64 halfs
            const u16* g = &Wsel[(size_t)(nBase + bn) * DMODEL + k0 + bc];
            ASYNC_LDS_B128(bL, g, 0);
            ASYNC_LDS_B128(bL, g, 16);
            ASYNC_LDS_B128(bL, g, 32);
            ASYNC_LDS_B128(bL, g, 48);
        }
        WAIT_ASYNCCNT0();
        __syncthreads();

        #pragma unroll
        for (int kk = 0; kk < 64; kk += 32) {
            v16bf af[2], bfr[2];
            #pragma unroll
            for (int i = 0; i < 2; ++i) {
                int r = waveM + i * 16 + ln;
                af[i] = make_frag(*(const v4u*)&As[r][kk + hs * 8],
                                  *(const v4u*)&As[r][kk + 16 + hs * 8]);
            }
            #pragma unroll
            for (int j = 0; j < 2; ++j) {
                int n = waveN + j * 16 + ln;
                bfr[j] = make_frag(*(const v4u*)&Bs[n][kk + hs * 8],
                                   *(const v4u*)&Bs[n][kk + 16 + hs * 8]);
            }
            #pragma unroll
            for (int i = 0; i < 2; ++i)
                #pragma unroll
                for (int j = 0; j < 2; ++j)
                    acc[i][j] = WMMA_BF16(af[i], bfr[j], acc[i][j]);
        }
        __syncthreads();
    }

    #pragma unroll
    for (int i = 0; i < 2; ++i)
        #pragma unroll
        for (int j = 0; j < 2; ++j)
            #pragma unroll
            for (int r = 0; r < 8; ++r) {
                int m = mBase + waveM + i * 16 + hs * 8 + r;   // b*256 + t
                int n = nBase + waveN + j * 16 + ln;           // h*64 + d
                int b = m >> 8, t = m & 255;
                int h = n >> 6, d = n & 63;
                int hb = h * BATCH + b;
                u16 val = f2bf(acc[i][j][r]);
                if (w == 0)
                    Q[(size_t)hb * (TLEN * DHEAD) + t * DHEAD + d] = val;
                else if (w == 1)
                    Kh[(size_t)hb * (TLEN * DHEAD) + t * DHEAD + d] = val;
                else
                    Vt[(size_t)hb * (TLEN * DHEAD) + d * TLEN + t] = val;
            }
}

// ---------------------------------------------------------------------------
// Kernel 4: scores[hb][t][s] = (q . k) / 16, masked NEG_BIG where s <= t
// per hb: 256x256 = q(256x64) @ k^T ; WG tile 128x128, wave 32x64 (2x4)
// ---------------------------------------------------------------------------
__global__ __launch_bounds__(256)
void scores_gemm(const u16* __restrict__ Q, const u16* __restrict__ Kh,
                 float* __restrict__ S) {
    const int hb    = blockIdx.z;
    const int tBase = blockIdx.y * 128;
    const int sBase = blockIdx.x * 128;
    const int tid   = threadIdx.x;
    const int lane  = tid & 31;
    const int wave  = tid >> 5;
    const int hs    = lane >> 4;
    const int ln    = lane & 15;
    const int waveM = (wave >> 1) * 32;   // 0..96
    const int waveN = (wave & 1) * 64;    // 0 / 64

    const u16* q = Q  + (size_t)hb * (TLEN * DHEAD);
    const u16* k = Kh + (size_t)hb * (TLEN * DHEAD);
    v8f acc[2][4] = {};

    #pragma unroll
    for (int k0 = 0; k0 < DHEAD; k0 += 32) {
        v16bf af[2], bfr[4];
        #pragma unroll
        for (int i = 0; i < 2; ++i) {
            int m = tBase + waveM + i * 16 + ln;
            const u16* p = &q[(size_t)m * DHEAD + k0];
            af[i] = make_frag(*(const v4u*)&p[hs * 8], *(const v4u*)&p[16 + hs * 8]);
        }
        #pragma unroll
        for (int j = 0; j < 4; ++j) {
            int n = sBase + waveN + j * 16 + ln;
            const u16* p = &k[(size_t)n * DHEAD + k0];
            bfr[j] = make_frag(*(const v4u*)&p[hs * 8], *(const v4u*)&p[16 + hs * 8]);
        }
        #pragma unroll
        for (int i = 0; i < 2; ++i)
            #pragma unroll
            for (int j = 0; j < 4; ++j)
                acc[i][j] = WMMA_BF16(af[i], bfr[j], acc[i][j]);
    }

    const float scale = 0.0625f;   // 1/sqrt(T)
    #pragma unroll
    for (int i = 0; i < 2; ++i)
        #pragma unroll
        for (int j = 0; j < 4; ++j)
            #pragma unroll
            for (int r = 0; r < 8; ++r) {
                int t = tBase + waveM + i * 16 + hs * 8 + r;
                int s = sBase + waveN + j * 16 + ln;
                float v = (s <= t) ? NEG_BIG : acc[i][j][r] * scale;
                S[(size_t)hb * (TLEN * TLEN) + (size_t)t * TLEN + s] = v;
            }
}

// ---------------------------------------------------------------------------
// Kernel 5: softmax over the hb axis (axis 0 of [HB, T, T]); bf16 probs out.
// One WG per t; thread owns one s; online max/sum over 1024 hb values.
// Masked columns (all NEG_BIG) naturally produce uniform 1/1024.
// ---------------------------------------------------------------------------
__global__ __launch_bounds__(256)
void softmax_hb(const float* __restrict__ S, u16* __restrict__ P) {
    const int t = blockIdx.x;
    const int s = threadIdx.x;
    const size_t base = (size_t)t * TLEN + s;
    float m = -3.4e38f, l = 0.0f;
    for (int hb = 0; hb < HB; ++hb) {
        __builtin_prefetch(&S[base + (size_t)(hb + 8) * (TLEN * TLEN)], 0, 1);
        float x  = S[base + (size_t)hb * (TLEN * TLEN)];
        float mn = fmaxf(m, x);
        l = l * __expf(m - mn) + __expf(x - mn);
        m = mn;
    }
    const float inv = 1.0f / l;
    for (int hb = 0; hb < HB; ++hb) {
        float x = S[base + (size_t)hb * (TLEN * TLEN)];
        P[base + (size_t)hb * (TLEN * TLEN)] = f2bf(__expf(x - m) * inv);
    }
}

// ---------------------------------------------------------------------------
// Kernel 6: out[b][t][h*64+d] = sum_s probs[hb][t][s] * v[hb][s][d]
// per hb: C(256x64) = P(256x256) @ Vt-as-B ; WG tile 128x64, wave 32x32 (2x2)
// ---------------------------------------------------------------------------
__global__ __launch_bounds__(256)
void out_gemm(const u16* __restrict__ P, const u16* __restrict__ Vt,
              float* __restrict__ out) {
    const int hb    = blockIdx.z;
    const int tBase = blockIdx.y * 128;
    const int tid   = threadIdx.x;
    const int lane  = tid & 31;
    const int wave  = tid >> 5;
    const int hs    = lane >> 4;
    const int ln    = lane & 15;
    const int waveM = (wave >> 1) * 32;   // 0..96
    const int waveN = (wave & 1) * 32;    // 0 / 32

    const u16* p = P  + (size_t)hb * (TLEN * TLEN);
    const u16* v = Vt + (size_t)hb * (TLEN * DHEAD);
    v8f acc[2][2] = {};

    #pragma unroll
    for (int k0 = 0; k0 < TLEN; k0 += 32) {
        v16bf af[2], bfr[2];
        #pragma unroll
        for (int i = 0; i < 2; ++i) {
            int m = tBase + waveM + i * 16 + ln;
            const u16* ap = &p[(size_t)m * TLEN + k0];
            af[i] = make_frag(*(const v4u*)&ap[hs * 8], *(const v4u*)&ap[16 + hs * 8]);
        }
        #pragma unroll
        for (int j = 0; j < 2; ++j) {
            int n = waveN + j * 16 + ln;   // n = d
            const u16* bp = &v[(size_t)n * TLEN + k0];
            bfr[j] = make_frag(*(const v4u*)&bp[hs * 8], *(const v4u*)&bp[16 + hs * 8]);
        }
        #pragma unroll
        for (int i = 0; i < 2; ++i)
            #pragma unroll
            for (int j = 0; j < 2; ++j)
                acc[i][j] = WMMA_BF16(af[i], bfr[j], acc[i][j]);
    }

    const int h = hb >> 6, b = hb & 63;
    #pragma unroll
    for (int i = 0; i < 2; ++i)
        #pragma unroll
        for (int j = 0; j < 2; ++j)
            #pragma unroll
            for (int r = 0; r < 8; ++r) {
                int t = tBase + waveM + i * 16 + hs * 8 + r;
                int d = waveN + j * 16 + ln;
                out[((size_t)(b * TLEN + t)) * DMODEL + h * DHEAD + d] = acc[i][j][r];
            }
}

// ---------------------------------------------------------------------------
extern "C" void kernel_launch(void* const* d_in, const int* in_sizes, int n_in,
                              void* d_out, int out_size, void* d_ws, size_t ws_size,
                              hipStream_t stream) {
    const float* x    = (const float*)d_in[0];
    const float* pos  = (const float*)d_in[1];
    const float* Wq   = (const float*)d_in[2];
    const float* Wk   = (const float*)d_in[3];
    const float* Wv   = (const float*)d_in[4];
    float* out = (float*)d_out;

    // workspace carve-up (bf16 unless noted)
    u16* Wt  = (u16*)d_ws;                             // [3][1024][1024] (n,k)
    u16* Tmp = Wt  + (size_t)3 * DMODEL * DMODEL;      // [16384][1024]
    u16* Q   = Tmp + (size_t)BATCH * TLEN * DMODEL;    // [1024][256][64]
    u16* Kh  = Q   + (size_t)HB * TLEN * DHEAD;        // [1024][256][64]
    u16* Vt  = Kh  + (size_t)HB * TLEN * DHEAD;        // [1024][64][256]
    float* S = (float*)(Vt + (size_t)HB * TLEN * DHEAD);   // [1024][256][256] f32
    u16* P   = (u16*)(S + (size_t)HB * TLEN * TLEN);       // [1024][256][256]

    prep_weights<<<dim3((3 * DMODEL * DMODEL) / 256), dim3(256), 0, stream>>>(Wq, Wk, Wv, Wt);
    prep_temporal<<<dim3((BATCH * TLEN * DMODEL) / (256 * 4)), dim3(256), 0, stream>>>(x, pos, Tmp);
    qkv_gemm<<<dim3(DMODEL / 128, (BATCH * TLEN) / 64, 3), dim3(256), 0, stream>>>(Tmp, Wt, Q, Kh, Vt);
    scores_gemm<<<dim3(TLEN / 128, TLEN / 128, HB), dim3(256), 0, stream>>>(Q, Kh, S);
    softmax_hb<<<dim3(TLEN), dim3(256), 0, stream>>>(S, P);
    out_gemm<<<dim3(1, TLEN / 128, HB), dim3(256), 0, stream>>>(P, Vt, out);
}
